// MPNNGNN_43293270344035
// MI455X (gfx1250) — matrix operand order repaired and checked by reading, not compiled, not run on hardware
//
#include <hip/hip_runtime.h>
#include <hip/hip_bf16.h>
#include <math.h>

// ---------------------------------------------------------------------------
// MPNN-GNN for MI455X (gfx1250, wave32, WMMA).
//  * only 21 distinct edge-conditioned W matrices -> precompute table (42KB)
//  * H == 32 == K of V_WMMA_F32_16X16X32_BF16 -> one WMMA per 16-edge group
//    per 16-col half; GRU gemms are 12 WMMAs per 16-node group.
//  * dst indices broadcast via ds_bpermute (__shfl) instead of 16 dependent
//    global loads per lane in the scatter loop.
//  * bf16 packing via v_cvt_pk_bf16_f32 when available.
// ---------------------------------------------------------------------------

typedef __bf16 v16bf __attribute__((ext_vector_type(16)));
typedef float  v8f   __attribute__((ext_vector_type(8)));

union BF16x16 {
  v16bf v;
  unsigned short u[16];
  unsigned int   d[8];
};

__device__ __forceinline__ unsigned short f2bf(float f) {
  unsigned int b = __float_as_uint(f);
  unsigned int r = b + 0x7FFFu + ((b >> 16) & 1u);   // round-to-nearest-even
  return (unsigned short)(r >> 16);
}

#if __has_builtin(__builtin_amdgcn_cvt_pk_bf16_f32)
typedef __bf16 v2bf __attribute__((ext_vector_type(2)));
__device__ __forceinline__ unsigned int f2bf2(float a, float b) {
  union { v2bf v; unsigned int u; } cv;
  cv.v = __builtin_amdgcn_cvt_pk_bf16_f32(a, b);
  return cv.u;
}
#else
__device__ __forceinline__ unsigned int f2bf2(float a, float b) {
  return (unsigned int)f2bf(a) | ((unsigned int)f2bf(b) << 16);
}
#endif

__device__ __forceinline__ float sigmoidf_(float x) { return 1.0f / (1.0f + expf(-x)); }

static constexpr int H        = 32;
static constexpr int NODE_IN  = 64;
static constexpr int EDGE_IN  = 16;
static constexpr int EDGE_HID = 128;
static constexpr int H2       = H * H;

// ---------------------------------------------------------------------------
// Prep: per-edge-id MLP -> W table, packed straight into bf16 WMMA B-layout.
// B layout (16x16x32): e<8 -> K=8*(lane/16)+e, e>=8 -> K=16+8*(lane/16)+(e-8); N=lane%16.
// ---------------------------------------------------------------------------
__global__ void k_prep_edge_w(const float* __restrict__ edge_emb,
                              const float* __restrict__ eW1, const float* __restrict__ eb1,
                              const float* __restrict__ eW2, const float* __restrict__ eb2,
                              unsigned short* __restrict__ Wpk) {
  __shared__ float x[EDGE_HID];
  __shared__ float Wsh[H2];
  int id = blockIdx.x;
  int t  = threadIdx.x;
  if (t < EDGE_HID) {
    const float* er = edge_emb + id * EDGE_IN;
    float acc = eb1[t];
#pragma unroll
    for (int j = 0; j < EDGE_IN; j++) acc = fmaf(er[j], eW1[j * EDGE_HID + t], acc);
    x[t] = acc > 0.0f ? acc : 0.0f;
  }
  __syncthreads();
  for (int o = t; o < H2; o += blockDim.x) {
    float acc = eb2[o];
    for (int k = 0; k < EDGE_HID; k++) acc = fmaf(x[k], eW2[k * H2 + o], acc);
    Wsh[o] = acc;          // Wsh[k*H + n] = W[k][n], msg[n] = sum_k h[k]*W[k][n]
  }
  __syncthreads();
  if (t < 64) {
    int half = t >> 5, lane = t & 31;
    int n  = (lane & 15) + 16 * half;
    int kb = (lane & 16) ? 8 : 0;
#pragma unroll
    for (int e = 0; e < 16; e++)
      Wpk[(((size_t)id * 2 + half) * 32 + lane) * 16 + e] = f2bf(Wsh[(kb + (e < 8 ? e : e + 8)) * H + n]);
  }
}

// Pack GRU weights (B = W^T, i.e. B[k][o] = W[o][k]) into B-layout tiles.
// 12 tiles: [ih t=0..5 | hh t=0..5], tile t covers output cols 16t..16t+15.
__global__ void k_prep_gru_pack(const float* __restrict__ Wih,
                                const float* __restrict__ Whh,
                                unsigned short* __restrict__ Gpk) {
  int b = blockIdx.x, lane = threadIdx.x;
  int mat = b / 6, t = b % 6;
  const float* W = mat ? Whh : Wih;             // [96][32] row-major
  int o  = 16 * t + (lane & 15);
  int kb = (lane & 16) ? 8 : 0;
#pragma unroll
  for (int e = 0; e < 16; e++)
    Gpk[((size_t)b * 32 + lane) * 16 + e] = f2bf(W[o * H + kb + (e < 8 ? e : e + 8)]);
}

// ---------------------------------------------------------------------------
// Node projection: h0 = relu(node_emb[node_ids] @ proj_W + proj_b).
// 343 distinct embedding rows -> fully L1/L2 resident.
// ---------------------------------------------------------------------------
__global__ void k_node_proj(const int* __restrict__ node_ids,
                            const float* __restrict__ emb,
                            const float* __restrict__ W, const float* __restrict__ b,
                            float* __restrict__ h, int N) {
  int idx = blockIdx.x * blockDim.x + threadIdx.x;
  if (idx >= N * H) return;
  int n = idx >> 5, c = idx & 31;
  const float* er = emb + (size_t)node_ids[n] * NODE_IN;
  float acc = b[c];
#pragma unroll 8
  for (int k = 0; k < NODE_IN; k++) acc = fmaf(er[k], W[k * H + c], acc);
  h[idx] = acc > 0.0f ? acc : 0.0f;
}

// ---------------------------------------------------------------------------
// Deterministic counting sort of edges by edge_id (ballot-rank, no fp order
// dependence). 21 bins, one wave each.
// ---------------------------------------------------------------------------
__global__ void k_count(const int* __restrict__ eid, int E, int* __restrict__ counts) {
  int id = blockIdx.x, lane = threadIdx.x;
  int cnt = 0;
  for (int b = 0; b < E; b += 32) {
    int i = b + lane;
    bool mt = (i < E) && (eid[i] == id);
    unsigned long long mk = __ballot(mt);
    cnt += (int)__popcll(mk);
  }
  if (lane == 0) counts[id] = cnt;
}

__global__ void k_prefix(const int* __restrict__ counts, int nid,
                         int* __restrict__ offs, int* __restrict__ goffs) {
  if (threadIdx.x == 0 && blockIdx.x == 0) {
    int off = 0, goff = 0;
    for (int i = 0; i < nid; i++) {
      offs[i] = off; goffs[i] = goff;
      off += counts[i]; goff += (counts[i] + 15) >> 4;
    }
    offs[nid] = off; goffs[nid] = goff;
  }
}

__global__ void k_scatter(const int* __restrict__ eid, int E,
                          const int* __restrict__ offs, int* __restrict__ perm) {
  int id = blockIdx.x, lane = threadIdx.x;
  int cur = offs[id];
  unsigned long long below = (1ull << lane) - 1ull;
  for (int b = 0; b < E; b += 32) {
    int i = b + lane;
    bool mt = (i < E) && (eid[i] == id);
    unsigned long long mk = __ballot(mt);
    if (mt) perm[cur + (int)__popcll(mk & below)] = i;
    cur += (int)__popcll(mk);
  }
}

__global__ void k_init_agg(float* __restrict__ agg, const float* __restrict__ bias, int NH) {
  int i = blockIdx.x * blockDim.x + threadIdx.x;
  if (i < NH) agg[i] = bias[i & 31];
}

// ---------------------------------------------------------------------------
// Edge messages: one wave = 16 same-id edges. A = h[src] rows (16x32 bf16,
// ISA 16-bit A layout), B = packed W[id] halves. D scattered with fp atomics.
// dst row indices are loaded once by lanes 0..15 and broadcast via __shfl
// (ds_bpermute) -- no global loads in the scatter loop.
// EXEC is all-ones at the WMMAs (wave-uniform exit only; clamps are selects).
// ---------------------------------------------------------------------------
__global__ void k_edge_msg(const float* __restrict__ hcur,
                           const unsigned short* __restrict__ Wpk,
                           const int* __restrict__ perm,
                           const int* __restrict__ src, const int* __restrict__ dst,
                           const int* __restrict__ offs, const int* __restrict__ goffs,
                           int nid, float* __restrict__ agg) {
  int wid  = (blockIdx.x * blockDim.x + threadIdx.x) >> 5;
  int lane = threadIdx.x & 31;
  int total = goffs[nid];
  if (wid >= total) return;                       // uniform per wave
  int id = 0;
  while (id < nid - 1 && goffs[id + 1] <= wid) id++;
  int chunk = wid - goffs[id];
  int base  = offs[id] + chunk * 16;
  int cnt   = offs[id + 1] - base; cnt = cnt > 16 ? 16 : cnt;

  // A gather: lane owns row m = lane%16; lane-half selects K-chunks.
  int m  = lane & 15;
  int em = (m < cnt) ? m : 0;                     // clamp padding rows
  int e  = perm[base + em];
  int srow = src[e];
  int drow = dst[e];                              // row m's destination
  const float* hr = hcur + (size_t)srow * H;
  int kb = (lane & 16) ? 8 : 0;
  BF16x16 A;
#pragma unroll
  for (int j = 0; j < 4; j++) {
    A.d[j]     = f2bf2(hr[kb + 2 * j],      hr[kb + 2 * j + 1]);
    A.d[4 + j] = f2bf2(hr[16 + kb + 2 * j], hr[16 + kb + 2 * j + 1]);
  }
  const v16bf* bp = reinterpret_cast<const v16bf*>(Wpk);
  v16bf B0 = bp[((size_t)id * 2 + 0) * 32 + lane];
  v16bf B1 = bp[((size_t)id * 2 + 1) * 32 + lane];

  // Broadcast the 8 dst rows this lane will scatter (rows rb..rb+7).
  int rb = (lane & 16) ? 8 : 0;
  int drows[8];
#pragma unroll
  for (int r = 0; r < 8; r++) drows[r] = __shfl(drow, rb + r);

  v8f z = {};
  v8f c0 = __builtin_amdgcn_wmma_f32_16x16x32_bf16(false, A.v, false, B0, (short)0, z, false, false);
  v8f c1 = __builtin_amdgcn_wmma_f32_16x16x32_bf16(false, A.v, false, B1, (short)0, z, false, false);

  // D layout: reg r -> M = rb + r, N = lane%16 (+16 for half 1).
  int n = lane & 15;
#pragma unroll
  for (int r = 0; r < 8; r++) {
    int M = rb + r;
    if (M < cnt) {
      float* ap = agg + (size_t)drows[r] * H + n;
      atomicAdd(ap,      c0[r]);
      atomicAdd(ap + 16, c1[r]);
    }
  }
}

// ---------------------------------------------------------------------------
// GRU step: one wave = 16 nodes. m = relu(agg) and hidden form two A tiles;
// 12 WMMAs produce gi/gh (96 cols = 6 tiles each); gates fused in-register.
// Lane holds cols c and c+16 of the 32 outputs for its 8 rows.
// ---------------------------------------------------------------------------
__global__ void k_gru(const float* __restrict__ agg, const float* __restrict__ hcur,
                      float* __restrict__ hnext, const unsigned short* __restrict__ Gpk,
                      const float* __restrict__ bih, const float* __restrict__ bhh,
                      int N, int ngroups) {
  int g = (blockIdx.x * blockDim.x + threadIdx.x) >> 5;
  if (g >= ngroups) return;                       // uniform per wave
  int lane = threadIdx.x & 31;
  int m  = lane & 15;
  int kb = (lane & 16) ? 8 : 0;
  int rowm = g * 16 + m; if (rowm >= N) rowm = N - 1;
  const float* ar = agg  + (size_t)rowm * H;
  const float* hr = hcur + (size_t)rowm * H;
  BF16x16 Am, Ah;
#pragma unroll
  for (int j = 0; j < 4; j++) {
    float a0 = ar[kb + 2 * j];          a0 = a0 > 0.0f ? a0 : 0.0f;
    float a1 = ar[kb + 2 * j + 1];      a1 = a1 > 0.0f ? a1 : 0.0f;
    float a2 = ar[16 + kb + 2 * j];     a2 = a2 > 0.0f ? a2 : 0.0f;
    float a3 = ar[16 + kb + 2 * j + 1]; a3 = a3 > 0.0f ? a3 : 0.0f;
    Am.d[j]     = f2bf2(a0, a1);
    Am.d[4 + j] = f2bf2(a2, a3);
    Ah.d[j]     = f2bf2(hr[kb + 2 * j],      hr[kb + 2 * j + 1]);
    Ah.d[4 + j] = f2bf2(hr[16 + kb + 2 * j], hr[16 + kb + 2 * j + 1]);
  }
  const v16bf* bp = reinterpret_cast<const v16bf*>(Gpk);
  v8f gi[6], gh[6];
#pragma unroll
  for (int t = 0; t < 6; t++) {
    v8f z = {};
    gi[t] = __builtin_amdgcn_wmma_f32_16x16x32_bf16(false, Am.v, false, bp[(size_t)t       * 32 + lane], (short)0, z, false, false);
    gh[t] = __builtin_amdgcn_wmma_f32_16x16x32_bf16(false, Ah.v, false, bp[(size_t)(6 + t) * 32 + lane], (short)0, z, false, false);
  }
  int rb = (lane & 16) ? 8 : 0;
  int c = lane & 15, c2 = c + 16;
  float biR0 = bih[c],  biZ0 = bih[32 + c],  biN0 = bih[64 + c];
  float biR1 = bih[c2], biZ1 = bih[32 + c2], biN1 = bih[64 + c2];
  float bhR0 = bhh[c],  bhZ0 = bhh[32 + c],  bhN0 = bhh[64 + c];
  float bhR1 = bhh[c2], bhZ1 = bhh[32 + c2], bhN1 = bhh[64 + c2];
#pragma unroll
  for (int r = 0; r < 8; r++) {
    int row = g * 16 + rb + r;
    if (row >= N) continue;
    float hp0 = hcur[(size_t)row * H + c];
    float hp1 = hcur[(size_t)row * H + c2];
    float rg0 = sigmoidf_(gi[0][r] + biR0 + gh[0][r] + bhR0);
    float zg0 = sigmoidf_(gi[2][r] + biZ0 + gh[2][r] + bhZ0);
    float nn0 = tanhf(gi[4][r] + biN0 + rg0 * (gh[4][r] + bhN0));
    float rg1 = sigmoidf_(gi[1][r] + biR1 + gh[1][r] + bhR1);
    float zg1 = sigmoidf_(gi[3][r] + biZ1 + gh[3][r] + bhZ1);
    float nn1 = tanhf(gi[5][r] + biN1 + rg1 * (gh[5][r] + bhN1));
    hnext[(size_t)row * H + c]  = (1.0f - zg0) * nn0 + zg0 * hp0;
    hnext[(size_t)row * H + c2] = (1.0f - zg1) * nn1 + zg1 * hp1;
  }
}

// ---------------------------------------------------------------------------
extern "C" void kernel_launch(void* const* d_in, const int* in_sizes, int n_in,
                              void* d_out, int out_size, void* d_ws, size_t ws_size,
                              hipStream_t stream) {
  (void)n_in; (void)out_size; (void)ws_size;
  const int*   node_ids = (const int*)d_in[0];
  const int*   edge_ids = (const int*)d_in[1];
  const int*   src      = (const int*)d_in[2];
  const int*   dst      = (const int*)d_in[3];
  const float* node_emb = (const float*)d_in[4];
  const float* edge_emb = (const float*)d_in[5];
  const float* proj_W   = (const float*)d_in[6];
  const float* proj_b   = (const float*)d_in[7];
  const float* eW1      = (const float*)d_in[8];
  const float* eb1      = (const float*)d_in[9];
  const float* eW2      = (const float*)d_in[10];
  const float* eb2      = (const float*)d_in[11];
  const float* conv_b   = (const float*)d_in[12];
  const float* gWih     = (const float*)d_in[13];
  const float* gWhh     = (const float*)d_in[14];
  const float* gbih     = (const float*)d_in[15];
  const float* gbhh     = (const float*)d_in[16];

  const int N   = in_sizes[0];
  const int E   = in_sizes[1];
  const int NID = in_sizes[5] / EDGE_IN;          // 21 distinct edge ids

  auto al = [](size_t x) { return (x + 255) & ~(size_t)255; };
  char* ws = (char*)d_ws;
  size_t off = 0;
  float* hA  = (float*)(ws + off); off = al(off + (size_t)N * H * 4);
  float* hB  = (float*)(ws + off); off = al(off + (size_t)N * H * 4);
  float* agg = (float*)(ws + off); off = al(off + (size_t)N * H * 4);
  unsigned short* Wpk = (unsigned short*)(ws + off); off = al(off + (size_t)NID * 2 * 32 * 16 * 2);
  unsigned short* Gpk = (unsigned short*)(ws + off); off = al(off + (size_t)12 * 32 * 16 * 2);
  int* counts = (int*)(ws + off); off = al(off + (size_t)NID * 4);
  int* offs   = (int*)(ws + off); off = al(off + (size_t)(NID + 1) * 4);
  int* goffs  = (int*)(ws + off); off = al(off + (size_t)(NID + 1) * 4);
  int* perm   = (int*)(ws + off); off = al(off + (size_t)E * 4);

  // one-time prep (per launch; deterministic)
  k_prep_edge_w<<<NID, 256, 0, stream>>>(edge_emb, eW1, eb1, eW2, eb2, Wpk);
  k_prep_gru_pack<<<12, 32, 0, stream>>>(gWih, gWhh, Gpk);
  int nh = N * H;
  k_node_proj<<<(nh + 255) / 256, 256, 0, stream>>>(node_ids, node_emb, proj_W, proj_b, hA, N);
  k_count<<<NID, 32, 0, stream>>>(edge_ids, E, counts);
  k_prefix<<<1, 1, 0, stream>>>(counts, NID, offs, goffs);
  k_scatter<<<NID, 32, 0, stream>>>(edge_ids, E, offs, perm);

  int maxGroups  = (E + 15) / 16 + NID;           // upper bound incl. partial bins
  int edgeBlocks = (maxGroups + 7) / 8;           // 8 waves / 256-thread block
  int ngroups    = (N + 15) / 16;
  int gruBlocks  = (ngroups + 7) / 8;

  float* cur = hA;
  float* nxt = hB;
  for (int step = 0; step < 6; step++) {
    k_init_agg<<<(nh + 255) / 256, 256, 0, stream>>>(agg, conv_b, nh);
    k_edge_msg<<<edgeBlocks, 256, 0, stream>>>(cur, Wpk, perm, src, dst, offs, goffs, NID, agg);
    float* wout = (step == 5) ? (float*)d_out : nxt;
    k_gru<<<gruBlocks, 256, 0, stream>>>(agg, cur, wout, Gpk, gbih, gbhh, N, ngroups);
    float* tmp = cur; cur = nxt; nxt = tmp;
  }
}
